// StandardMoE_48361331752981
// MI455X (gfx1250) — matrix operand order repaired and checked by reading, not compiled, not run on hardware
//
#include <hip/hip_runtime.h>
#include <cstdint>

typedef unsigned short u16;
typedef unsigned int   u32;

typedef __bf16 bf16_t;
typedef bf16_t v16bf __attribute__((ext_vector_type(16)));
typedef float  v8f   __attribute__((ext_vector_type(8)));

#define D_MODEL   2048
#define D_FF      1408
#define N_EXPERTS 6
#define TOPK      2
#define BATCH     8
#define SEQ       4096

#define BM  128
#define BN  128
#define BK  32
#define LDT 40   // padded LDS row length in halves (80 B rows, 16B-aligned chunks)

__device__ __forceinline__ u16 f2bf(float f) {
  u32 u = __float_as_uint(f);
  u32 r = (u + 0x7FFFu + ((u >> 16) & 1u)) >> 16;  // round-to-nearest-even
  return (u16)r;
}

// Async DMA: 16B per lane, global -> LDS, tracked by ASYNCcnt (CDNA5 §10.7/§15.18).
// VDST = per-lane LDS byte address (low 32 bits of generic shared pointer),
// VADDR = per-lane 64-bit global address, saddr = off.
__device__ __forceinline__ void async_b128(const u16* g, u16* l) {
  u32 lds = (u32)(uintptr_t)l;
  asm volatile("global_load_async_to_lds_b128 %0, %1, off"
               :: "v"(lds), "v"(g) : "memory");
}
__device__ __forceinline__ void wait_async0() {
  asm volatile("s_wait_asynccnt 0x0" ::: "memory");
}

union FragU { uint4 q[2]; v16bf v; };

// ---------------- f32 -> bf16 bulk convert (4 elems/thread) ----------------
__global__ __launch_bounds__(256) void k_f32_to_bf16x4(const float4* __restrict__ in,
                                                       uint2* __restrict__ out,
                                                       long long n4) {
  long long i = (long long)blockIdx.x * 256 + threadIdx.x;
  if (i >= n4) return;
  float4 f = in[i];
  uint2 o;
  o.x = (u32)f2bf(f.x) | ((u32)f2bf(f.y) << 16);
  o.y = (u32)f2bf(f.z) | ((u32)f2bf(f.w) << 16);
  out[i] = o;
}

// ---------------- f32 [R][C] -> bf16 [C][R] tiled transpose (per expert) ----------------
__global__ __launch_bounds__(256) void k_transpose_bf16(const float* __restrict__ in,
                                                        u16* __restrict__ out,
                                                        int R, int C) {
  __shared__ u16 t[32][33];
  const int e = blockIdx.z;
  const float* src = in + (size_t)e * R * C;
  u16* dst = out + (size_t)e * R * C;
  const int r0 = blockIdx.x * 32, c0 = blockIdx.y * 32;
  const int tx = threadIdx.x & 31, ty = threadIdx.x >> 5;
#pragma unroll
  for (int i = ty; i < 32; i += 8)
    t[i][tx] = f2bf(src[(size_t)(r0 + i) * C + c0 + tx]);
  __syncthreads();
#pragma unroll
  for (int i = ty; i < 32; i += 8)
    dst[(size_t)(c0 + i) * R + r0 + tx] = t[tx][i];
}

// ---------------- deterministic mean over sequence ----------------
__global__ __launch_bounds__(256) void k_mean_part(const float* __restrict__ x,
                                                   float* __restrict__ part) {
  int d  = blockIdx.x * 256 + threadIdx.x;
  int sp = blockIdx.y;
  int b  = blockIdx.z;
  const float* p = x + ((size_t)b * SEQ + (size_t)sp * 256) * D_MODEL + d;
  float s = 0.f;
  for (int i = 0; i < 256; ++i) s += p[(size_t)i * D_MODEL];
  part[((size_t)b * 16 + sp) * D_MODEL + d] = s;
}

__global__ __launch_bounds__(256) void k_mean_reduce(const float* __restrict__ part,
                                                     float* __restrict__ meanx) {
  int d = blockIdx.x * 256 + threadIdx.x;
  int b = blockIdx.y;
  float s = 0.f;
  for (int i = 0; i < 16; ++i) s += part[((size_t)b * 16 + i) * D_MODEL + d];
  meanx[(size_t)b * D_MODEL + d] = s * (1.f / (float)SEQ);
}

// ---------------- gating: logits, top-2, renormalized softmax ----------------
__global__ __launch_bounds__(256) void k_gate(const float* __restrict__ meanx,
                                              const float* __restrict__ gate_W,
                                              int* __restrict__ gidx,
                                              float* __restrict__ gw) {
  __shared__ float red[256];
  __shared__ float logits[BATCH][N_EXPERTS];
  int tx = threadIdx.x;
  for (int be = 0; be < BATCH * N_EXPERTS; ++be) {
    int b = be / N_EXPERTS, e = be % N_EXPERTS;
    float p = 0.f;
    for (int i = tx; i < D_MODEL; i += 256)
      p += meanx[b * D_MODEL + i] * gate_W[e * D_MODEL + i];
    red[tx] = p;
    __syncthreads();
    for (int s = 128; s > 0; s >>= 1) {
      if (tx < s) red[tx] += red[tx + s];
      __syncthreads();
    }
    if (tx == 0) logits[b][e] = red[0];
    __syncthreads();
  }
  if (tx < BATCH) {
    int i0 = 0; float v0 = logits[tx][0];
    for (int e = 1; e < N_EXPERTS; ++e)
      if (logits[tx][e] > v0) { v0 = logits[tx][e]; i0 = e; }
    int i1 = (i0 == 0) ? 1 : 0; float v1 = logits[tx][i1];
    for (int e = 0; e < N_EXPERTS; ++e)
      if (e != i0 && logits[tx][e] > v1) { v1 = logits[tx][e]; i1 = e; }
    // renormalized top-2 softmax: full denominator cancels
    float t = expf(v1 - v0);
    float s = 1.f + t;
    gidx[tx * 2 + 0] = i0;  gidx[tx * 2 + 1] = i1;
    gw[tx * 2 + 0] = 1.f / s;  gw[tx * 2 + 1] = t / s;
  }
}

// ---------------- GEMM1: h = w_k * gelu(x @ W1[e] + b1[e])  (bf16 out) ----------------
// A = xh [S][D] bf16; B = W1t [F][D] bf16 (pre-transposed: row n holds k-contiguous data)
__global__ __launch_bounds__(256) void k_gemm1(
    const u16* __restrict__ xh, const u16* __restrict__ W1t,
    const float* __restrict__ b1, const int* __restrict__ gidx,
    const float* __restrict__ gw, u16* __restrict__ hbuf) {
  const int m0 = blockIdx.x * BM;
  const int n0 = blockIdx.y * BN;
  const int bk = blockIdx.z;                 // b*2 + k
  const int b  = bk >> 1, kk = bk & 1;
  const int e  = gidx[b * 2 + kk];
  const float wk = gw[b * 2 + kk];

  const u16* __restrict__ A  = xh  + (size_t)b * SEQ * D_MODEL;      // [S][D]
  const u16* __restrict__ Bt = W1t + (size_t)e * D_FF * D_MODEL;     // [F][D]

  __shared__ __align__(16) u16 As[2][BM * LDT];
  __shared__ __align__(16) u16 Bs[2][BN * LDT];

  const int tid = threadIdx.x;
  const int lane = tid & 31, wv = tid >> 5;
  const int wave_m = wv >> 2, wave_n = wv & 3;     // 2 x 4 wave grid
  const int lr = lane & 15, lh = lane >> 4;

  const int r0c = tid >> 2,          ac = (tid & 3) * 8;   // tile: 2 x 16B chunks/thread
  const int r1c = (tid + 256) >> 2;

  auto issue_tile = [&](int kt, int buf) {
    async_b128(A  + (size_t)(m0 + r0c) * D_MODEL + kt * BK + ac, &As[buf][r0c * LDT + ac]);
    async_b128(A  + (size_t)(m0 + r1c) * D_MODEL + kt * BK + ac, &As[buf][r1c * LDT + ac]);
    async_b128(Bt + (size_t)(n0 + r0c) * D_MODEL + kt * BK + ac, &Bs[buf][r0c * LDT + ac]);
    async_b128(Bt + (size_t)(n0 + r1c) * D_MODEL + kt * BK + ac, &Bs[buf][r1c * LDT + ac]);
  };

  v8f acc[4][2];
#pragma unroll
  for (int i = 0; i < 4; ++i)
#pragma unroll
    for (int j = 0; j < 2; ++j)
      acc[i][j] = v8f{0.f, 0.f, 0.f, 0.f, 0.f, 0.f, 0.f, 0.f};

  const int NK = D_MODEL / BK;   // 64
  issue_tile(0, 0);
  wait_async0();
  __syncthreads();
#pragma unroll 1
  for (int kt = 0; kt < NK; ++kt) {
    const int cur = kt & 1;
    const bool more = (kt + 1) < NK;
    if (more) issue_tile(kt + 1, cur ^ 1);
    FragU af[4], bf[2];
#pragma unroll
    for (int i = 0; i < 4; ++i) {   // A frag: lane<16 k{0..7,16..23}, lane>=16 k{8..15,24..31}
      const u16* base = &As[cur][(wave_m * 64 + i * 16 + lr) * LDT];
      af[i].q[0] = *(const uint4*)(base + lh * 8);
      af[i].q[1] = *(const uint4*)(base + 16 + lh * 8);
    }
#pragma unroll
    for (int j = 0; j < 2; ++j) {   // B frag: lane<16 k{0..7,16..23}, lane>=16 k{8..15,24..31} (col = lr)
      const u16* base = &Bs[cur][(wave_n * 32 + j * 16 + lr) * LDT];
      bf[j].q[0] = *(const uint4*)(base + lh * 8);
      bf[j].q[1] = *(const uint4*)(base + 16 + lh * 8);
    }
#pragma unroll
    for (int i = 0; i < 4; ++i)
#pragma unroll
      for (int j = 0; j < 2; ++j)
        acc[i][j] = __builtin_amdgcn_wmma_f32_16x16x32_bf16(
            false, af[i].v, false, bf[j].v, (short)0, acc[i][j], false, false);
    if (more) wait_async0();
    __syncthreads();
  }

  // epilogue: bias + exact-erf gelu, fold in gate weight, store bf16
  u16* __restrict__ H = hbuf + (size_t)bk * SEQ * D_FF;
  const int rbase = m0 + wave_m * 64 + (lh ? 8 : 0);
  const int cb    = n0 + wave_n * 32 + lr;
#pragma unroll
  for (int i = 0; i < 4; ++i)
#pragma unroll
    for (int j = 0; j < 2; ++j) {
      const int col = cb + j * 16;
      const float bias = b1[e * D_FF + col];
#pragma unroll
      for (int v = 0; v < 8; ++v) {
        const int row = rbase + i * 16 + v;
        float xv = acc[i][j][v] + bias;
        xv = 0.5f * xv * (1.f + erff(xv * 0.70710678118654752440f));
        H[(size_t)row * D_FF + col] = f2bf(xv * wk);
      }
    }
}

// ---------------- GEMM2: out = sum_k h_k @ W2[e_k]  + sum_k w_k*b2[e_k] ----------------
// A = hbuf [S][F] bf16 (gate weight folded in); B = W2t [D][F] bf16 (pre-transposed)
__global__ __launch_bounds__(256) void k_gemm2(
    const u16* __restrict__ hbuf, const u16* __restrict__ W2t,
    const float* __restrict__ b2, const int* __restrict__ gidx,
    const float* __restrict__ gw, float* __restrict__ out) {
  const int m0 = blockIdx.x * BM;
  const int n0 = blockIdx.y * BN;
  const int b  = blockIdx.z;
  const int e0 = gidx[b * 2 + 0], e1 = gidx[b * 2 + 1];
  const float w0 = gw[b * 2 + 0], w1 = gw[b * 2 + 1];

  __shared__ __align__(16) u16 As[2][BM * LDT];
  __shared__ __align__(16) u16 Bs[2][BN * LDT];

  const int tid = threadIdx.x;
  const int lane = tid & 31, wv = tid >> 5;
  const int wave_m = wv >> 2, wave_n = wv & 3;
  const int lr = lane & 15, lh = lane >> 4;

  const int r0c = tid >> 2,          ac = (tid & 3) * 8;
  const int r1c = (tid + 256) >> 2;

  const int NK2   = D_FF / BK;   // 44
  const int NSTEP = 2 * NK2;     // both experts in one accumulation chain

  auto issue_tile = [&](int step, int buf) {
    const int kkk = (step >= NK2) ? 1 : 0;
    const int kt  = step - kkk * NK2;
    const u16* Ap = hbuf + (size_t)(b * 2 + kkk) * SEQ * D_FF;          // [S][F]
    const u16* Bp = W2t + (size_t)(kkk ? e1 : e0) * D_MODEL * D_FF;     // [D][F]
    async_b128(Ap + (size_t)(m0 + r0c) * D_FF + kt * BK + ac, &As[buf][r0c * LDT + ac]);
    async_b128(Ap + (size_t)(m0 + r1c) * D_FF + kt * BK + ac, &As[buf][r1c * LDT + ac]);
    async_b128(Bp + (size_t)(n0 + r0c) * D_FF + kt * BK + ac, &Bs[buf][r0c * LDT + ac]);
    async_b128(Bp + (size_t)(n0 + r1c) * D_FF + kt * BK + ac, &Bs[buf][r1c * LDT + ac]);
  };

  v8f acc[4][2];
#pragma unroll
  for (int i = 0; i < 4; ++i)
#pragma unroll
    for (int j = 0; j < 2; ++j)
      acc[i][j] = v8f{0.f, 0.f, 0.f, 0.f, 0.f, 0.f, 0.f, 0.f};

  issue_tile(0, 0);
  wait_async0();
  __syncthreads();
#pragma unroll 1
  for (int st = 0; st < NSTEP; ++st) {
    const int cur = st & 1;
    const bool more = (st + 1) < NSTEP;
    if (more) issue_tile(st + 1, cur ^ 1);
    FragU af[4], bf[2];
#pragma unroll
    for (int i = 0; i < 4; ++i) {
      const u16* base = &As[cur][(wave_m * 64 + i * 16 + lr) * LDT];
      af[i].q[0] = *(const uint4*)(base + lh * 8);
      af[i].q[1] = *(const uint4*)(base + 16 + lh * 8);
    }
#pragma unroll
    for (int j = 0; j < 2; ++j) {
      const u16* base = &Bs[cur][(wave_n * 32 + j * 16 + lr) * LDT];
      bf[j].q[0] = *(const uint4*)(base + lh * 8);
      bf[j].q[1] = *(const uint4*)(base + 16 + lh * 8);
    }
#pragma unroll
    for (int i = 0; i < 4; ++i)
#pragma unroll
      for (int j = 0; j < 2; ++j)
        acc[i][j] = __builtin_amdgcn_wmma_f32_16x16x32_bf16(
            false, af[i].v, false, bf[j].v, (short)0, acc[i][j], false, false);
    if (more) wait_async0();
    __syncthreads();
  }

  const int rbase = m0 + wave_m * 64 + (lh ? 8 : 0);
  const int cb    = n0 + wave_n * 32 + lr;
#pragma unroll
  for (int i = 0; i < 4; ++i)
#pragma unroll
    for (int j = 0; j < 2; ++j) {
      const int col = cb + j * 16;
      const float bias = w0 * b2[e0 * D_MODEL + col] + w1 * b2[e1 * D_MODEL + col];
#pragma unroll
      for (int v = 0; v < 8; ++v) {
        const int row = rbase + i * 16 + v;
        out[(size_t)(b * SEQ + row) * D_MODEL + col] = acc[i][j][v] + bias;
      }
    }
}

// ---------------- launch ----------------
extern "C" void kernel_launch(void* const* d_in, const int* in_sizes, int n_in,
                              void* d_out, int out_size, void* d_ws, size_t ws_size,
                              hipStream_t stream) {
  (void)in_sizes; (void)n_in; (void)out_size; (void)ws_size;
  const float* x      = (const float*)d_in[0];
  const float* gate_W = (const float*)d_in[1];
  const float* W1     = (const float*)d_in[2];
  const float* b1     = (const float*)d_in[3];
  const float* W2     = (const float*)d_in[4];
  const float* b2     = (const float*)d_in[5];
  float* out = (float*)d_out;

  char* ws = (char*)d_ws;
  size_t off = 0;
  auto wsalloc = [&](size_t bytes) -> char* {
    char* p = ws + off;
    off = (off + bytes + 255) & ~(size_t)255;
    return p;
  };

  u16*   xh    = (u16*)wsalloc((size_t)BATCH * SEQ * D_MODEL * 2);          // 134 MB
  u16*   W1t   = (u16*)wsalloc((size_t)N_EXPERTS * D_MODEL * D_FF * 2);     // 34.6 MB  [E][F][D]
  u16*   W2t   = (u16*)wsalloc((size_t)N_EXPERTS * D_FF * D_MODEL * 2);     // 34.6 MB  [E][D][F]
  u16*   hbuf  = (u16*)wsalloc((size_t)BATCH * TOPK * SEQ * D_FF * 2);      // 176 MB
  float* part  = (float*)wsalloc((size_t)BATCH * 16 * D_MODEL * 4);
  float* meanx = (float*)wsalloc((size_t)BATCH * D_MODEL * 4);
  int*   gidx  = (int*)wsalloc(BATCH * TOPK * sizeof(int));
  float* gwv   = (float*)wsalloc(BATCH * TOPK * sizeof(float));

  {
    long long n4 = (long long)BATCH * SEQ * D_MODEL / 4;                    // 16,777,216
    k_f32_to_bf16x4<<<(unsigned)(n4 / 256), 256, 0, stream>>>((const float4*)x, (uint2*)xh, n4);
  }
  // W1 [E][D][F] f32 -> W1t [E][F][D] bf16 ; W2 [E][F][D] f32 -> W2t [E][D][F] bf16
  k_transpose_bf16<<<dim3(D_MODEL / 32, D_FF / 32, N_EXPERTS), 256, 0, stream>>>(W1, W1t, D_MODEL, D_FF);
  k_transpose_bf16<<<dim3(D_FF / 32, D_MODEL / 32, N_EXPERTS), 256, 0, stream>>>(W2, W2t, D_FF, D_MODEL);

  k_mean_part  <<<dim3(D_MODEL / 256, 16, BATCH), 256, 0, stream>>>(x, part);
  k_mean_reduce<<<dim3(D_MODEL / 256, BATCH),     256, 0, stream>>>(part, meanx);
  k_gate       <<<1, 256, 0, stream>>>(meanx, gate_W, gidx, gwv);

  k_gemm1<<<dim3(SEQ / BM, D_FF / BN, BATCH * TOPK), 256, 0, stream>>>(xh, W1t, b1, gidx, gwv, hbuf);
  k_gemm2<<<dim3(SEQ / BM, D_MODEL / BN, BATCH),     256, 0, stream>>>(hbuf, W2t, b2, gidx, gwv, out);
}
